// GIN_36481452212846
// MI455X (gfx1250) — matrix operand order repaired and checked by reading, not compiled, not run on hardware
//
#include <hip/hip_runtime.h>

typedef __attribute__((ext_vector_type(16))) __bf16 v16bf;
typedef __attribute__((ext_vector_type(8)))  __bf16 v8bf;
typedef __attribute__((ext_vector_type(8)))  float  v8f;
typedef __attribute__((ext_vector_type(4)))  float  f32x4;

#define GIN_N_NODES 50000
#define GIN_N_EDGES 800000

// ---------------------------------------------------------------------------
// Pre-swizzle an f32 weight matrix W[K][64] into bf16 B-fragments for
// V_WMMA_F32_16X16X32_BF16. Fragment element order:
//   frag[((ct*nks + ks)*32 + lane)*16 + j]
// where for B (32x16): b[j] = W[ks*32 + (lane>=16 ? 8 : 0) + (j&7) + (j>=8 ? 16 : 0)][ct*16 + (lane&15)]
// ---------------------------------------------------------------------------
__global__ void gin_prep_frag(const float* __restrict__ W,
                              unsigned short* __restrict__ frag, int nks) {
  int idx = blockIdx.x * blockDim.x + threadIdx.x;
  int total = nks * 4 * 32 * 16;
  if (idx >= total) return;
  int j    = idx & 15;
  int lane = (idx >> 4) & 31;
  int t    = idx >> 9;          // ct*nks + ks
  int ks   = t % nks;
  int ct   = t / nks;
  int col  = ct * 16 + (lane & 15);
  int kk   = ks * 32 + ((lane >> 4) << 3) + (j & 7) + ((j >> 3) << 4);
  __bf16 b = (__bf16)W[kk * 64 + col];
  frag[idx] = __builtin_bit_cast(unsigned short, b);
}

__global__ void gin_zero4(float* __restrict__ p, int n4) {
  int i = blockIdx.x * blockDim.x + threadIdx.x;
  if (i < n4) {
    f32x4 z = {0.f, 0.f, 0.f, 0.f};
    ((f32x4*)p)[i] = z;
  }
}

// ---------------------------------------------------------------------------
// Edge scatter: agg[dst[e]][:] += h[src[e]][:]   (float atomics, land in L2)
// One thread per (edge, 4-float chunk).
// ---------------------------------------------------------------------------
__global__ void gin_scatter(const float* __restrict__ h, const int* __restrict__ src,
                            const int* __restrict__ dst, float* __restrict__ agg, int d) {
  long tid = (long)blockIdx.x * blockDim.x + threadIdx.x;
  int cpe = d >> 2;
  long total = (long)GIN_N_EDGES * cpe;
  if (tid >= total) return;
  int e  = (int)(tid / cpe);
  int ch = ((int)(tid % cpe)) << 2;
  int s = src[e], t = dst[e];
  f32x4 v = *(const f32x4*)(h + (size_t)s * d + ch);
  float* p = agg + (size_t)t * d + ch;
  atomicAdd(p + 0, v[0]);
  atomicAdd(p + 1, v[1]);
  atomicAdd(p + 2, v[2]);
  atomicAdd(p + 3, v[3]);
}

// ---------------------------------------------------------------------------
// Fused MLP: z2 = relu((h+agg) @ W1 + b1) @ W2 + b2, plus per-column
// sum / sum-of-squares for BatchNorm. One wave = 16 rows x 64 cols.
// ---------------------------------------------------------------------------
template <int K>
__global__ __launch_bounds__(128) void gin_mlp_wmma(
    const float* __restrict__ hin, const float* __restrict__ agg,
    const unsigned short* __restrict__ fragW1, const float* __restrict__ b1,
    const unsigned short* __restrict__ fragW2, const float* __restrict__ b2,
    float* __restrict__ zout, float* __restrict__ stats) {
  __shared__ __align__(16) __bf16 ldsY[4][16][64];

  const int lane    = threadIdx.x & 31;
  const int wave    = threadIdx.x >> 5;
  const int rowbase = blockIdx.x * 64 + wave * 16;
  if (rowbase >= GIN_N_NODES) return;  // uniform per wave (N % 16 == 0); no barriers used

  const int l15   = lane & 15;
  const int half8 = (lane >> 4) << 3;  // 0 for lanes 0-15, 8 for lanes 16-31
  const int nk    = K >> 5;

  const int r = rowbase + l15;
  const float* ph = hin + (size_t)r * K;
  const float* pa = agg + (size_t)r * K;

  const v8f vzero = {0.f, 0.f, 0.f, 0.f, 0.f, 0.f, 0.f, 0.f};
  v8f acc[4];
#pragma unroll
  for (int ct = 0; ct < 4; ++ct) acc[ct] = vzero;

  // ---- GEMM1: (h+agg)[16xK] @ W1[Kx64] ----
  // A fragment (16-bit, 16x32): lane<16 -> K {kb..kb+7, kb+16..kb+23} with kb=k0;
  // lane>=16 -> kb=k0+8.
#pragma unroll
  for (int ks = 0; ks < nk; ++ks) {
    const int kb = ks * 32 + half8;
    f32x4 x0 = *(const f32x4*)(ph + kb)      + *(const f32x4*)(pa + kb);
    f32x4 x1 = *(const f32x4*)(ph + kb + 4)  + *(const f32x4*)(pa + kb + 4);
    f32x4 x2 = *(const f32x4*)(ph + kb + 16) + *(const f32x4*)(pa + kb + 16);
    f32x4 x3 = *(const f32x4*)(ph + kb + 20) + *(const f32x4*)(pa + kb + 20);
    v16bf a;
#pragma unroll
    for (int i = 0; i < 4; ++i) {
      a[i]      = (__bf16)x0[i];
      a[4 + i]  = (__bf16)x1[i];
      a[8 + i]  = (__bf16)x2[i];
      a[12 + i] = (__bf16)x3[i];
    }
#pragma unroll
    for (int ct = 0; ct < 4; ++ct) {
      v16bf b = *(const v16bf*)(fragW1 + (((ct * nk + ks) * 32 + lane) << 4));
      acc[ct] = __builtin_amdgcn_wmma_f32_16x16x32_bf16(
          false, a, false, b, (short)0, acc[ct], false, false);
    }
  }

  // ---- bias + ReLU, stash Y1 tile in LDS (bf16, row-major 16x64) ----
  // C layout: lane<16 -> (M=j, N=l15+16ct); lane>=16 -> (M=j+8, N=l15+16ct)
#pragma unroll
  for (int ct = 0; ct < 4; ++ct) {
    const int c = ct * 16 + l15;
    const float bb = b1[c];
#pragma unroll
    for (int j = 0; j < 8; ++j) {
      float y = acc[ct][j] + bb;
      y = y > 0.f ? y : 0.f;
      ldsY[wave][j + half8][c] = (__bf16)y;
    }
  }

  // ---- GEMM2: Y1[16x64] @ W2[64x64] (A re-gathered from LDS) ----
  v8f acc2[4];
#pragma unroll
  for (int ct = 0; ct < 4; ++ct) acc2[ct] = vzero;

#pragma unroll
  for (int ks = 0; ks < 2; ++ks) {
    const int kb = ks * 32 + half8;
    const __bf16* py = &ldsY[wave][l15][0];
    v8bf lo = *(const v8bf*)(py + kb);       // 16B LDS read
    v8bf hi = *(const v8bf*)(py + kb + 16);  // 16B LDS read
    v16bf a;
#pragma unroll
    for (int i = 0; i < 8; ++i) { a[i] = lo[i]; a[8 + i] = hi[i]; }
#pragma unroll
    for (int ct = 0; ct < 4; ++ct) {
      v16bf b = *(const v16bf*)(fragW2 + (((ct * 2 + ks) * 32 + lane) << 4));
      acc2[ct] = __builtin_amdgcn_wmma_f32_16x16x32_bf16(
          false, a, false, b, (short)0, acc2[ct], false, false);
    }
  }

  // ---- bias, store pre-BN z, accumulate column sum / sum^2 ----
#pragma unroll
  for (int ct = 0; ct < 4; ++ct) {
    const int c = ct * 16 + l15;
    const float bb = b2[c];
    float s = 0.f, q = 0.f;
#pragma unroll
    for (int j = 0; j < 8; ++j) {
      float z = acc2[ct][j] + bb;
      zout[(size_t)(rowbase + j + half8) * 64 + c] = z;
      s += z;
      q += z * z;
    }
    // combine lane pair (L, L^16): both hold the same column c
    s += __shfl_xor(s, 16, 32);
    q += __shfl_xor(q, 16, 32);
    if (lane < 16) {
      atomicAdd(&stats[c], s);
      atomicAdd(&stats[64 + c], q);
    }
  }
}

__global__ void gin_bn_finalize(const float* __restrict__ stats,
                                const float* __restrict__ gamma,
                                const float* __restrict__ beta,
                                float* __restrict__ scsh) {
  int c = threadIdx.x;
  if (c >= 64) return;
  const float invN = 1.0f / (float)GIN_N_NODES;
  float mean = stats[c] * invN;
  float var  = stats[64 + c] * invN - mean * mean;
  float sc   = gamma[c] * rsqrtf(var + 1e-5f);
  scsh[c]      = sc;
  scsh[64 + c] = beta[c] - mean * sc;
}

__global__ void gin_bn_apply(float* __restrict__ z, const float* __restrict__ scsh) {
  int i = blockIdx.x * blockDim.x + threadIdx.x;
  if (i >= GIN_N_NODES * 16) return;  // 50000*64/4 float4 groups
  int c4 = (i & 15) << 2;
  f32x4 v = ((f32x4*)z)[i];
#pragma unroll
  for (int k = 0; k < 4; ++k) {
    float y = v[k] * scsh[c4 + k] + scsh[64 + c4 + k];
    v[k] = y > 0.f ? y : 0.f;
  }
  ((f32x4*)z)[i] = v;
}

// ---------------------------------------------------------------------------
extern "C" void kernel_launch(void* const* d_in, const int* in_sizes, int n_in,
                              void* d_out, int out_size, void* d_ws, size_t ws_size,
                              hipStream_t stream) {
  (void)in_sizes; (void)n_in; (void)out_size; (void)ws_size;
  const float* h    = (const float*)d_in[0];
  const int*   src  = (const int*)d_in[1];
  const int*   dst  = (const int*)d_in[2];
  const float* W1_0 = (const float*)d_in[3];
  const float* b1_0 = (const float*)d_in[4];
  const float* W2_0 = (const float*)d_in[5];
  const float* b2_0 = (const float*)d_in[6];
  const float* g_0  = (const float*)d_in[7];
  const float* be_0 = (const float*)d_in[8];
  const float* W1s  = (const float*)d_in[9];
  const float* b1s  = (const float*)d_in[10];
  const float* W2s  = (const float*)d_in[11];
  const float* b2s  = (const float*)d_in[12];
  const float* gs   = (const float*)d_in[13];
  const float* bes  = (const float*)d_in[14];

  // --- carve workspace (all sub-offsets 32B-aligned) ---
  char* w = (char*)d_ws;
  float* agg   = (float*)w; w += (size_t)GIN_N_NODES * 128 * 4;  // 25.6 MB
  float* hA    = (float*)w; w += (size_t)GIN_N_NODES * 64 * 4;   // 12.8 MB
  float* hB    = (float*)w; w += (size_t)GIN_N_NODES * 64 * 4;   // 12.8 MB
  float* stats = (float*)w; w += 512;   // colsum[64] | colsumsq[64]
  float* scsh  = (float*)w; w += 512;   // scale[64]  | shift[64]
  unsigned short* f1[4];
  unsigned short* f2[4];
  f1[0] = (unsigned short*)w; w += 4 * 4 * 32 * 16 * 2;  // K=128 fragments, 16 KB
  f2[0] = (unsigned short*)w; w += 2 * 4 * 32 * 16 * 2;  // 8 KB
  for (int i = 1; i < 4; ++i) {
    f1[i] = (unsigned short*)w; w += 2 * 4 * 32 * 16 * 2;
    f2[i] = (unsigned short*)w; w += 2 * 4 * 32 * 16 * 2;
  }

  // --- weight prep (deterministic, every call) ---
  gin_prep_frag<<<32, 256, 0, stream>>>(W1_0, f1[0], 4);
  gin_prep_frag<<<16, 256, 0, stream>>>(W2_0, f2[0], 2);
  for (int i = 0; i < 3; ++i) {
    gin_prep_frag<<<16, 256, 0, stream>>>(W1s + (size_t)i * 64 * 64, f1[i + 1], 2);
    gin_prep_frag<<<16, 256, 0, stream>>>(W2s + (size_t)i * 64 * 64, f2[i + 1], 2);
  }

  const float* hin = h;
  float* houts[4] = {hA, hB, hA, (float*)d_out};
  const int nblk = (GIN_N_NODES + 63) / 64;  // 782

  for (int L = 0; L < 4; ++L) {
    const int K = (L == 0) ? 128 : 64;
    const float* b1 = (L == 0) ? b1_0 : b1s + (size_t)(L - 1) * 64;
    const float* b2 = (L == 0) ? b2_0 : b2s + (size_t)(L - 1) * 64;
    const float* ga = (L == 0) ? g_0  : gs  + (size_t)(L - 1) * 64;
    const float* be = (L == 0) ? be_0 : bes + (size_t)(L - 1) * 64;
    float* hout = houts[L];

    const int n4 = GIN_N_NODES * K / 4;
    gin_zero4<<<(n4 + 255) / 256, 256, 0, stream>>>(agg, n4);
    gin_zero4<<<1, 64, 0, stream>>>(stats, 32);

    const long sc_total = (long)GIN_N_EDGES * (K / 4);
    gin_scatter<<<(int)((sc_total + 255) / 256), 256, 0, stream>>>(hin, src, dst, agg, K);

    if (K == 128)
      gin_mlp_wmma<128><<<nblk, 128, 0, stream>>>(hin, agg, f1[L], b1, f2[L], b2, hout, stats);
    else
      gin_mlp_wmma<64><<<nblk, 128, 0, stream>>>(hin, agg, f1[L], b1, f2[L], b2, hout, stats);

    gin_bn_finalize<<<1, 64, 0, stream>>>(stats, ga, be, scsh);
    gin_bn_apply<<<(GIN_N_NODES * 16 + 255) / 256, 256, 0, stream>>>(hout, scsh);

    hin = hout;
  }
}
